// ContrastiveLoss_76209899700384
// MI455X (gfx1250) — matrix operand order repaired and checked by reading, not compiled, not run on hardware
//
#include <hip/hip_runtime.h>
#include <hip/hip_bf16.h>

typedef __attribute__((ext_vector_type(16))) _Float16 v16h;
typedef __attribute__((ext_vector_type(8)))  _Float16 v8h;
typedef __attribute__((ext_vector_type(8)))  float    v8f;
typedef __attribute__((ext_vector_type(4)))  int      v4i;

#define NI 1024   // images
#define NO 36     // regions per image  (N dim, padded to 48 = 3 tiles)
#define NC 1024   // captions
#define NW 60     // words per caption  (M dim, padded to 64 = 4 tiles)
#define ND 64     // feature dim (K)    = 2 x K32 WMMA steps
#define IMGS_PER_WG 4
#define CAPS_PER_WG 64
#define MARGIN 0.2f
#define EPSV 1e-6f
#define NEGINF (-3.0e38f)

#if defined(__has_builtin)
# if __has_builtin(__builtin_amdgcn_global_load_async_to_lds_b128)
#  define HAVE_ASYNC_LDS 1
# endif
# if __has_builtin(__builtin_amdgcn_s_wait_asynccnt)
#  define HAVE_WAIT_ASYNC 1
# endif
#endif
#ifndef HAVE_ASYNC_LDS
# define HAVE_ASYNC_LDS 0
#endif
#ifndef HAVE_WAIT_ASYNC
# define HAVE_WAIT_ASYNC 0
#endif

// ---------------- f32 -> f16 conversion ----------------
__global__ void cvt_f32_f16_kernel(const float* __restrict__ x,
                                   _Float16* __restrict__ y, int n) {
  int idx = blockIdx.x * blockDim.x + threadIdx.x;
  if (idx < n) y[idx] = (_Float16)x[idx];
}

// Stage one caption (NW*ND f16 = 7680 B = 480 x 16B chunks) into LDS.
// Uniform 4 async-load instructions per wave: rounds 0..2 cover chunks
// 0..383, round 3 covers 384..479 with tid>=96 redundantly re-writing
// chunks 384..415 (identical data; benign) so ASYNCcnt is exactly 4.
__device__ __forceinline__ void stage_caption(const _Float16* __restrict__ src,
                                              _Float16* __restrict__ dst,
                                              int tid) {
#if HAVE_ASYNC_LDS
  __attribute__((address_space(1))) char* gp =
      (__attribute__((address_space(1))) char*)src;
  __attribute__((address_space(3))) char* lp =
      (__attribute__((address_space(3))) char*)dst;
#pragma unroll
  for (int k = 0; k < 3; ++k) {
    int off = (tid + k * 128) * 16;
    __builtin_amdgcn_global_load_async_to_lds_b128(
        (__attribute__((address_space(1))) v4i*)(gp + off),
        (__attribute__((address_space(3))) v4i*)(lp + off), 0, 0);
  }
  {
    int c3 = 384 + ((tid < 96) ? tid : (tid - 96));
    int off = c3 * 16;
    __builtin_amdgcn_global_load_async_to_lds_b128(
        (__attribute__((address_space(1))) v4i*)(gp + off),
        (__attribute__((address_space(3))) v4i*)(lp + off), 0, 0);
  }
#else
#pragma unroll
  for (int k = 0; k < 3; ++k) {
    int c = tid + k * 128;
    *(v8h*)(dst + c * 8) = *(const v8h*)(src + c * 8);
  }
  {
    int c3 = 384 + ((tid < 96) ? tid : (tid - 96));
    *(v8h*)(dst + c3 * 8) = *(const v8h*)(src + c3 * 8);
  }
#endif
}

__device__ __forceinline__ void wait_async_le4() {
#if HAVE_ASYNC_LDS
# if HAVE_WAIT_ASYNC
  __builtin_amdgcn_s_wait_asynccnt(4);
# else
  asm volatile("s_wait_asynccnt 0x4" ::: "memory");
# endif
#endif
}
__device__ __forceinline__ void wait_async_0() {
#if HAVE_ASYNC_LDS
# if HAVE_WAIT_ASYNC
  __builtin_amdgcn_s_wait_asynccnt(0);
# else
  asm volatile("s_wait_asynccnt 0x0" ::: "memory");
# endif
#endif
}

// ---------------- main cross-attention score kernel ----------------
// 4 waves per workgroup; wave w owns image blockIdx.x*4+w (its im tile is
// register-resident as the WMMA B operand). Captions stream through a
// double-buffered LDS stage shared by all 4 waves.
// Per caption: D(words x regions) = s_c (A, 16x32 tiles from LDS) x im^T
// (B, 32x16 tiles); masked max over words (M: in-lane + xor16), sum over
// valid regions (N: xor1..8), normalize, store score.
__global__ __launch_bounds__(128)
void xattn_scores_kernel(const _Float16* __restrict__ imh,
                         const _Float16* __restrict__ sh,
                         const int* __restrict__ im_l,
                         const int* __restrict__ s_l,
                         float* __restrict__ scores) {
  __shared__ __align__(16) _Float16 smem[2][NW * ND];

  const int tid  = threadIdx.x;
  const int wave = tid >> 5;
  const int lane = tid & 31;
  const int g    = lane >> 4;
  const int l16  = lane & 15;
  const int i    = blockIdx.x * IMGS_PER_WG + wave;   // this wave's image
  const int c0   = blockIdx.y * CAPS_PER_WG;

  // ---- resident B fragments: im[i]^T, 3 region tiles x 2 K-halves ----
  v16h bfrag[3][2];
  const _Float16* imbase = imh + (size_t)i * NO * ND;
#pragma unroll
  for (int nt = 0; nt < 3; ++nt) {
    int o = nt * 16 + l16;
    if (o >= NO) o = NO - 1;            // pad regions (excluded from sum)
    const _Float16* rowp = imbase + o * ND;
#pragma unroll
    for (int kh = 0; kh < 2; ++kh)
      bfrag[nt][kh] = *(const v16h*)(rowp + kh * 32 + g * 16);
  }
  const float denom = (float)im_l[i] + EPSV;

  // ---- prologue: stage first caption ----
  stage_caption(sh + (size_t)c0 * NW * ND, smem[0], tid);

  for (int cc = 0; cc < CAPS_PER_WG; ++cc) {
    const int c   = c0 + cc;
    const int cur = cc & 1;
    if (cc + 1 < CAPS_PER_WG) {
      stage_caption(sh + (size_t)(c + 1) * NW * ND, smem[cur ^ 1], tid);
      wait_async_le4();                 // current buffer's loads done
    } else {
      wait_async_0();
    }
    __syncthreads();                    // all waves' stages complete

    const int clen = s_l[c];
    const _Float16* sm = smem[cur];

    float runmax0 = NEGINF, runmax1 = NEGINF, runmax2 = NEGINF;
#pragma unroll
    for (int tw = 0; tw < 4; ++tw) {    // word tiles (M)
      int w = tw * 16 + l16;
      int wc = (w < NW) ? w : (NW - 1);
      const _Float16* rp = sm + wc * ND;
      // A fragments from LDS (16x32 f16 layout)
      v16h a0, a1;
      {
        v8h lo0 = *(const v8h*)(rp + g * 16);
        v8h hi0 = *(const v8h*)(rp + g * 16 + 8);
        v8h lo1 = *(const v8h*)(rp + 32 + g * 16);
        v8h hi1 = *(const v8h*)(rp + 32 + g * 16 + 8);
#pragma unroll
        for (int k = 0; k < 8; ++k) {
          a0[k] = lo0[k]; a0[8 + k] = hi0[k];
          a1[k] = lo1[k]; a1[8 + k] = hi1[k];
        }
      }
      const int wbase = tw * 16 + 8 * g;  // word index of acc[j=0]
      float rm[3];
#pragma unroll
      for (int nt = 0; nt < 3; ++nt) {  // region tiles (N)
        v8f acc = {};
        acc = __builtin_amdgcn_wmma_f32_16x16x32_f16(
            false, a0, false, bfrag[nt][0], (short)0, acc, false, false);
        acc = __builtin_amdgcn_wmma_f32_16x16x32_f16(
            false, a1, false, bfrag[nt][1], (short)0, acc, false, false);
        // masked in-lane max over this tile's 8 word rows
        float m = NEGINF;
#pragma unroll
        for (int j = 0; j < 8; ++j) {
          float v = ((wbase + j) < clen) ? acc[j] : NEGINF;
          m = fmaxf(m, v);
        }
        rm[nt] = m;
      }
      runmax0 = fmaxf(runmax0, rm[0]);
      runmax1 = fmaxf(runmax1, rm[1]);
      runmax2 = fmaxf(runmax2, rm[2]);
    }

    // combine the two half-wave word groups (one shuffle per region tile)
    runmax0 = fmaxf(runmax0, __shfl_xor(runmax0, 16, 32));
    runmax1 = fmaxf(runmax1, __shfl_xor(runmax1, 16, 32));
    runmax2 = fmaxf(runmax2, __shfl_xor(runmax2, 16, 32));

    // sum over valid regions: lane l16 holds regions {l16, 16+l16, 32+l16}
    float contrib = runmax0 + runmax1 + ((l16 < 4) ? runmax2 : 0.0f);
    contrib += __shfl_xor(contrib, 1, 32);
    contrib += __shfl_xor(contrib, 2, 32);
    contrib += __shfl_xor(contrib, 4, 32);
    contrib += __shfl_xor(contrib, 8, 32);

    if (lane == 0) scores[(size_t)i * NC + c] = contrib / denom;

    __syncthreads();                    // everyone done reading buf[cur]
  }
}

// ---------------- loss epilogue (deterministic, no float atomics) ----------------
__global__ void diag_kernel(const float* __restrict__ scores,
                            float* __restrict__ diagv) {
  int i = blockIdx.x * blockDim.x + threadIdx.x;
  if (i < NI) diagv[i] = scores[(size_t)i * NC + i];
}

__global__ __launch_bounds__(256)
void row_loss_kernel(const float* __restrict__ scores,
                     const float* __restrict__ diagv,
                     float* __restrict__ rowsum) {
  const int i = blockIdx.x;
  const float di = diagv[i];
  float acc = 0.0f;
  for (int c = threadIdx.x; c < NC; c += 256) {
    if (c != i) {
      float sc = scores[(size_t)i * NC + c];
      float t1 = MARGIN + sc - di;        // caption-retrieval hinge
      if (t1 > 0.0f) acc += t1;
      float t2 = MARGIN + sc - diagv[c];  // image-retrieval hinge
      if (t2 > 0.0f) acc += t2;
    }
  }
  __shared__ float smr[256];
  smr[threadIdx.x] = acc;
  __syncthreads();
  for (int s = 128; s > 0; s >>= 1) {
    if (threadIdx.x < s) smr[threadIdx.x] += smr[threadIdx.x + s];
    __syncthreads();
  }
  if (threadIdx.x == 0) rowsum[i] = smr[0];
}

__global__ __launch_bounds__(256)
void final_loss_kernel(const float* __restrict__ rowsum, float* __restrict__ out) {
  float acc = 0.0f;
  for (int i = threadIdx.x; i < NI; i += 256) acc += rowsum[i];
  __shared__ float smr[256];
  smr[threadIdx.x] = acc;
  __syncthreads();
  for (int s = 128; s > 0; s >>= 1) {
    if (threadIdx.x < s) smr[threadIdx.x] += smr[threadIdx.x + s];
    __syncthreads();
  }
  if (threadIdx.x == 0) out[0] = smr[0] / ((float)NI * (float)NC);
}

extern "C" void kernel_launch(void* const* d_in, const int* in_sizes, int n_in,
                              void* d_out, int out_size, void* d_ws, size_t ws_size,
                              hipStream_t stream) {
  const float* im   = (const float*)d_in[0];   // [NI, NO, ND] f32
  const float* s    = (const float*)d_in[1];   // [NC, NW, ND] f32
  const int*   im_l = (const int*)d_in[2];     // [NI]
  const int*   s_l  = (const int*)d_in[3];     // [NC]

  char* ws = (char*)d_ws;
  const size_t imh_bytes = (size_t)NI * NO * ND * sizeof(_Float16); //  4,718,592
  const size_t sh_bytes  = (size_t)NC * NW * ND * sizeof(_Float16); //  7,864,320
  const size_t sc_bytes  = (size_t)NI * NC * sizeof(float);         //  4,194,304
  _Float16* imh    = (_Float16*)(ws);
  _Float16* shh    = (_Float16*)(ws + imh_bytes);
  float*    scores = (float*)(ws + imh_bytes + sh_bytes);
  float*    diagv  = (float*)(ws + imh_bytes + sh_bytes + sc_bytes);
  float*    rowsum = (float*)(ws + imh_bytes + sh_bytes + sc_bytes + 4096);

  const int nim = NI * NO * ND;
  const int ns  = NC * NW * ND;
  cvt_f32_f16_kernel<<<(nim + 255) / 256, 256, 0, stream>>>(im, imh, nim);
  cvt_f32_f16_kernel<<<(ns + 255) / 256, 256, 0, stream>>>(s, shh, ns);

  dim3 grid(NI / IMGS_PER_WG, NC / CAPS_PER_WG);  // 256 x 16, 128-thread blocks
  xattn_scores_kernel<<<grid, 128, 0, stream>>>(imh, shh, im_l, s_l, scores);

  diag_kernel<<<(NI + 255) / 256, 256, 0, stream>>>(scores, diagv);
  row_loss_kernel<<<NI, 256, 0, stream>>>(scores, diagv, rowsum);
  final_loss_kernel<<<1, 256, 0, stream>>>(rowsum, (float*)d_out);
}
